// YOLOXHead_66855460930058
// MI455X (gfx1250) — matrix operand order repaired and checked by reading, not compiled, not run on hardware
//
#include <hip/hip_runtime.h>
#include <hip/hip_bf16.h>
#include <math.h>

// ---------------------------------------------------------------------------
// YOLOX head on MI455X (gfx1250): every conv is an implicit GEMM on
// v_wmma_f32_16x16x32_f16. Activations kept fp16 in [C][B*H*W] layout in the
// workspace; BN+SiLU fused in the GEMM epilogue; final 1x1 preds + decode.
// K staged 64-deep per barrier; all per-element addressing is shift/mask.
// ---------------------------------------------------------------------------

typedef __attribute__((ext_vector_type(16))) _Float16 v16h;
typedef __attribute__((ext_vector_type(8)))  _Float16 v8h;
typedef __attribute__((ext_vector_type(8)))  float    v8f;

#define BN_EPS_F 1e-5f

__device__ __forceinline__ float sigmoidf_(float x) {
    return 1.0f / (1.0f + __expf(-x));
}

// ---------------------------------------------------------------------------
// Generic implicit-GEMM conv.
//   TIN    : float (original NCHW inputs) or _Float16 ([C][BHW] activations)
//   KS     : 1 or 3 (SAME padding, stride 1)
//   FUSE   : true  -> y = SiLU(BN(conv)), fp16 out; false -> conv+bias, fp32
//   MGUARD : whether co < Cout guards are needed (pred GEMMs only)
// GEMM view: M = Cout, N = B*H*W, K = Cin*KS*KS (k = tap*Cin + ci).
// Block = 128 threads (4 waves), tile M=64 x N=64, K staged 64 per barrier.
// Wave w owns rows [16w,16w+16) x 64 cols -> 4 accums, 8 wmma per stage.
// ---------------------------------------------------------------------------
template <typename TIN, int KS, bool FUSE, bool MGUARD>
__global__ __launch_bounds__(128)
void conv_wmma_kernel(const TIN* __restrict__ X,
                      const float* __restrict__ Wt,     // [Cout,Cin,KS,KS] f32
                      const float* __restrict__ gamma,  // FUSE only
                      const float* __restrict__ bias,   // BN beta or pred bias
                      _Float16* __restrict__ Yh,
                      float* __restrict__ Yf,
                      int ciShift, int Cout, int B, int H, int W,
                      long strideC, long strideB)
{
    const int  Cin  = 1 << ciShift;
    const int  cMask = Cin - 1;
    const int  HW   = H * W;
    const long Ntot = (long)B * HW;
    const int  Ktot = Cin * KS * KS;
    const int  n0   = blockIdx.x * 64;
    const int  co0  = blockIdx.y * 64;
    const int  tid  = threadIdx.x;
    const int  lane = tid & 31;
    const int  wv   = tid >> 5;

    // Padded rows (72 halves = 144B) keep 16B alignment, spread LDS banks.
    constexpr int AP = 72, BP = 72;
    __shared__ __align__(16) _Float16 Asm[64 * AP];   // Asm[m][k]  (64x64)
    __shared__ __align__(16) _Float16 Bsm[64 * BP];   // Bsm[n][k]  (64x64, T)

    // ---- per-thread staging coordinates (invariant across K stages) ----
    // A: kk fixed, m strided;  B: nl fixed, kk strided.  (coalesced groups)
    const int kkA = tid & 63;
    const int mA0 = tid >> 6;            // 0 or 1
    const int nlB = tid & 63;
    const int kkB0 = tid >> 6;           // 0 or 1
    {
    }
    const long nB = n0 + nlB;
    const int  bB = (int)(nB / HW);      // once, not per stage
    const int  hwB = (int)(nB - (long)bB * HW);
    const int  hB = hwB / W;
    const int  wB = hwB - hB * W;
    const long baseB = (long)bB * strideB;

    v8f acc[4] = {};

    for (int k0 = 0; k0 < Ktot; k0 += 64) {
        // ---- stage A tile: 64 co x 64 k (f32 weights -> f16) ----
        {
            const int k  = k0 + kkA;
            const int r  = KS == 1 ? 0 : (k >> ciShift);
            const int ci = KS == 1 ? k : (k & cMask);
            // widx = (co*Cin+ci)*KS*KS + r   (3x3: ((..)*3+ky)*3+kx == ..*9+r)
            long widx = ((long)(co0 + mA0) * Cin + ci) * (KS * KS) + r;
            const long wstep = 2L * Cin * (KS * KS);
            // speculative prefetch of next stage's weight line
            if (k + 64 < Ktot) {
                const int kn  = k + 64;
                const int rn  = KS == 1 ? 0 : (kn >> ciShift);
                const int cin = KS == 1 ? kn : (kn & cMask);
                __builtin_prefetch(
                    Wt + ((long)(co0 + mA0) * Cin + cin) * (KS * KS) + rn, 0, 0);
            }
#pragma unroll 4
            for (int t = 0; t < 32; ++t) {
                const int m = mA0 + 2 * t;
                float v = 0.0f;
                if (!MGUARD || (co0 + m) < Cout) v = Wt[widx];
                Asm[m * AP + kkA] = (_Float16)v;
                widx += wstep;
            }
        }
        // ---- stage B tile (transposed): Bsm[n][k], im2col on the fly ----
        {
#pragma unroll 4
            for (int t = 0; t < 32; ++t) {
                const int kk = kkB0 + 2 * t;
                const int k  = k0 + kk;
                float v;
                if constexpr (KS == 1) {
                    v = (float)X[(long)k * strideC + baseB + hwB];
                } else {
                    const int r  = k >> ciShift;
                    const int ci = k & cMask;
                    const int ky = r / 3;              // const-div -> mul/shift
                    const int kx = r - ky * 3;
                    const int hh = hB + ky - 1;
                    const int ww = wB + kx - 1;
                    v = (hh >= 0 && hh < H && ww >= 0 && ww < W)
                            ? (float)X[(long)ci * strideC + baseB +
                                       (long)hh * W + ww]
                            : 0.0f;
                }
                Bsm[nlB * BP + kk] = (_Float16)v;
            }
        }
        __syncthreads();

        // ---- fragments per ISA 7.12.2 (wave32), two 32-K halves ----
        const int mr = lane & 15;
        const int hg = lane >> 4;
#pragma unroll
        for (int u = 0; u < 2; ++u) {
            const _Float16* arow = &Asm[(16 * wv + mr) * AP + 32 * u + hg * 8];
            v8h  alo = *(const v8h*)(arow);
            v8h  ahi = *(const v8h*)(arow + 16);
            v16h afrag = __builtin_shufflevector(alo, ahi,
                0, 1, 2, 3, 4, 5, 6, 7, 8, 9, 10, 11, 12, 13, 14, 15);
#pragma unroll
            for (int j = 0; j < 4; ++j) {
                const _Float16* brow =
                    &Bsm[(16 * j + mr) * BP + 32 * u + hg * 8];
                v8h  blo = *(const v8h*)(brow);
                v8h  bhi = *(const v8h*)(brow + 16);
                v16h bfrag = __builtin_shufflevector(blo, bhi,
                    0, 1, 2, 3, 4, 5, 6, 7, 8, 9, 10, 11, 12, 13, 14, 15);
                acc[j] = __builtin_amdgcn_wmma_f32_16x16x32_f16(
                    false, afrag, false, bfrag, (short)0, acc[j], false, false);
            }
        }
        __syncthreads();
    }

    // ---- epilogue: C/D layout: col = lane&15, row(r) = r + 8*(lane>>4) ----
#pragma unroll
    for (int j = 0; j < 4; ++j) {
        long n = n0 + 16 * j + (lane & 15);
#pragma unroll
        for (int r = 0; r < 8; ++r) {
            int co = co0 + 16 * wv + r + 8 * (lane >> 4);
            if (!MGUARD || co < Cout) {
                float v = acc[j][r];
                if constexpr (FUSE) {
                    float s = gamma[co] * rsqrtf(1.0f + BN_EPS_F);
                    v = s * v + bias[co];
                    v = v * sigmoidf_(v);                 // SiLU
                    Yh[(long)co * Ntot + n] = (_Float16)v;
                } else {
                    v += bias[co];
                    Yf[(long)co * Ntot + n] = v;
                }
            }
        }
    }
}

// ---------------------------------------------------------------------------
// Pack reg_pred (4x256) + obj_pred (1x256) into one M=5 GEMM operand.
// ---------------------------------------------------------------------------
__global__ void pack_regobj_kernel(const float* __restrict__ rw,
                                   const float* __restrict__ rb,
                                   const float* __restrict__ ow,
                                   const float* __restrict__ ob,
                                   float* __restrict__ Wp,
                                   float* __restrict__ Bp)
{
    int i = blockIdx.x * blockDim.x + threadIdx.x;
    if (i < 4 * 256)      Wp[i] = rw[i];
    else if (i < 5 * 256) Wp[i] = ow[i - 4 * 256];
    if (i < 4)            Bp[i] = rb[i];
    else if (i == 4)      Bp[4] = ob[0];
}

// ---------------------------------------------------------------------------
// Decode: out[b, level_off + hw, 0:85] from Yro (rows 0-3 reg, 4 obj) + Ycls.
// ---------------------------------------------------------------------------
__global__ void decode_kernel(const float* __restrict__ Yro,
                              const float* __restrict__ Ycls,
                              float* __restrict__ out,
                              int H, int W, int level_off, float stride_,
                              long Ntot)
{
    long idx = (long)blockIdx.x * blockDim.x + threadIdx.x;
    if (idx >= Ntot * 85) return;
    long n = idx / 85;
    int  c = (int)(idx - n * 85);
    int  HW = H * W;
    int  b  = (int)(n / HW);
    int  hw = (int)(n - (long)b * HW);
    int  h  = hw / W, w = hw - h * W;
    long orow = (long)b * 8400 + level_off + hw;
    float v;
    if (c < 4) {
        v = Yro[(long)c * Ntot + n];
        if (c == 0)      v = (v + (float)w) * stride_;
        else if (c == 1) v = (v + (float)h) * stride_;
        else             v = __expf(v) * stride_;
    } else if (c == 4) {
        v = sigmoidf_(Yro[4 * Ntot + n]);
    } else {
        v = sigmoidf_(Ycls[(long)(c - 5) * Ntot + n]);
    }
    out[orow * 85 + c] = v;
}

// ---------------------------------------------------------------------------
// Host-side orchestration.
// d_in order (setup_inputs() dict insertion order):
//   [0..2]  x0,x1,x2
//   then per level (21 entries): stem_w,stem_g,stem_b,
//     cls0_w,cls0_g,cls0_b, cls1_w,cls1_g,cls1_b,
//     reg0_w,reg0_g,reg0_b, reg1_w,reg1_g,reg1_b,
//     cls_pred_w,cls_pred_b, reg_pred_w,reg_pred_b, obj_pred_w,obj_pred_b
// ---------------------------------------------------------------------------
extern "C" void kernel_launch(void* const* d_in, const int* in_sizes, int n_in,
                              void* d_out, int out_size, void* d_ws, size_t ws_size,
                              hipStream_t stream)
{
    (void)in_sizes; (void)n_in; (void)out_size; (void)ws_size;

    const int  Bn = 8;
    const int  CSH[3]   = {8, 9, 10};     // log2(Cin): 256, 512, 1024
    const int  HS[3]    = {80, 40, 20};
    const int  STR[3]   = {8, 16, 32};
    const int  LOFF[3]  = {0, 6400, 8000};
    const long maxN     = 8L * 80 * 80;   // 51200 positions (level 0)

    // Workspace carve-out (fp16 activation ping-pong + pred outputs).
    char*  ws  = (char*)d_ws;
    size_t off = 0;
    auto carve = [&](size_t bytes) -> void* {
        void* p = ws + off;
        off = (off + bytes + 255) & ~(size_t)255;
        return p;
    };
    _Float16* S   = (_Float16*)carve(maxN * 256 * sizeof(_Float16)); // stem out
    _Float16* T1  = (_Float16*)carve(maxN * 256 * sizeof(_Float16));
    _Float16* T2  = (_Float16*)carve(maxN * 256 * sizeof(_Float16)); // cls final
    _Float16* T3  = (_Float16*)carve(maxN * 256 * sizeof(_Float16)); // reg final
    float*    Yro = (float*)carve(maxN * 16 * sizeof(float));        // reg+obj
    float*    Ycl = (float*)carve(maxN * 80 * sizeof(float));        // cls pred
    float*    Wp  = (float*)carve(16 * 256 * sizeof(float));
    float*    Bp  = (float*)carve(16 * sizeof(float));

    float* out = (float*)d_out;

    for (int lvl = 0; lvl < 3; ++lvl) {
        const int  csh = CSH[lvl];
        const int  Cin = 1 << csh;
        const int  H   = HS[lvl], W = HS[lvl];
        const long N   = (long)Bn * H * W;
        const int  pb  = 3 + lvl * 21;
        auto P = [&](int i) { return (const float*)d_in[pb + i]; };
        const float* x = (const float*)d_in[lvl];

        const dim3 blk(128);
        const dim3 g256((unsigned)(N / 64), 4);  // Cout=256 -> 4 M-tiles
        const dim3 g80((unsigned)(N / 64), 2);   // Cout=80
        const dim3 g16((unsigned)(N / 64), 1);   // Cout=5

        // stem: 1x1 Cin->256, fp32 NCHW input, BN+SiLU, fp16 [C][BHW] out
        conv_wmma_kernel<float, 1, true, false><<<g256, blk, 0, stream>>>(
            x, P(0), P(1), P(2), S, nullptr,
            csh, 256, Bn, H, W, (long)H * W, (long)Cin * H * W);

        // cls branch: two 3x3 256->256 (fp16 in/out)
        conv_wmma_kernel<_Float16, 3, true, false><<<g256, blk, 0, stream>>>(
            S, P(3), P(4), P(5), T1, nullptr, 8, 256, Bn, H, W, N, (long)H * W);
        conv_wmma_kernel<_Float16, 3, true, false><<<g256, blk, 0, stream>>>(
            T1, P(6), P(7), P(8), T2, nullptr, 8, 256, Bn, H, W, N, (long)H * W);

        // reg branch: two 3x3 256->256 (T1 reused after cls1 consumed it)
        conv_wmma_kernel<_Float16, 3, true, false><<<g256, blk, 0, stream>>>(
            S, P(9), P(10), P(11), T1, nullptr, 8, 256, Bn, H, W, N, (long)H * W);
        conv_wmma_kernel<_Float16, 3, true, false><<<g256, blk, 0, stream>>>(
            T1, P(12), P(13), P(14), T3, nullptr, 8, 256, Bn, H, W, N, (long)H * W);

        // cls_pred: 1x1 256->80, bias only, fp32 out
        conv_wmma_kernel<_Float16, 1, false, true><<<g80, blk, 0, stream>>>(
            T2, P(15), nullptr, P(16), nullptr, Ycl,
            8, 80, Bn, H, W, N, (long)H * W);

        // reg_pred + obj_pred packed into M=5 GEMM
        pack_regobj_kernel<<<(5 * 256 + 255) / 256, 256, 0, stream>>>(
            P(17), P(18), P(19), P(20), Wp, Bp);
        conv_wmma_kernel<_Float16, 1, false, true><<<g16, blk, 0, stream>>>(
            T3, Wp, nullptr, Bp, nullptr, Yro,
            8, 5, Bn, H, W, N, (long)H * W);

        // decode -> (B, 8400, 85) slice for this level
        long total = N * 85;
        decode_kernel<<<(unsigned)((total + 255) / 256), 256, 0, stream>>>(
            Yro, Ycl, out, H, W, LOFF[lvl], (float)STR[lvl], N);
    }
}
